// res_gnn_layer_35914516529843
// MI455X (gfx1250) — matrix operand order repaired and checked by reading, not compiled, not run on hardware
//
#include <hip/hip_runtime.h>

#define Bdim 2
#define Adim 5
#define Ndim 1000
#define Kdim 16
#define Fdim 256
#define Edim 3
#define KD   (Edim * Fdim)   // 768

typedef __attribute__((ext_vector_type(16))) __bf16 v16bf;
typedef __attribute__((ext_vector_type(8)))  float  v8f;

union Frag16 { unsigned int u[8]; v16bf v; };

__device__ __forceinline__ unsigned short f32_to_bf16_rne(float x) {
    unsigned int u = __float_as_uint(x);
    u += 0x7FFFu + ((u >> 16) & 1u);     // round-to-nearest-even
    return (unsigned short)(u >> 16);
}

// ---------------------------------------------------------------------------
// Kernel 1: fp32 -> bf16 weight conversion
// ---------------------------------------------------------------------------
__global__ void cvt_bf16_kernel(const float* __restrict__ src,
                                unsigned short* __restrict__ dst, int n) {
    int i = blockIdx.x * blockDim.x + threadIdx.x;
    if (i < n) dst[i] = f32_to_bf16_rne(src[i]);
}

// ---------------------------------------------------------------------------
// Kernel 2: per-(b,a,n) neighbor gather + per-etype aggregation
//   agg[b,a,n, e*F + f] = sum_{k: etype==e} nfeature[b,a,idx[k], f]   (bf16)
//   cntpack[b,a,n] = cnt0 | cnt1<<8 | cnt2<<16
// ---------------------------------------------------------------------------
__global__ __launch_bounds__(Fdim)
void agg_kernel(const float* __restrict__ nfeature,
                const int*   __restrict__ nn_idx,
                const int*   __restrict__ etype,
                unsigned short* __restrict__ agg_bf,
                unsigned int*   __restrict__ cntpack) {
    int n = blockIdx.x, a = blockIdx.y, b = blockIdx.z;
    int tid = threadIdx.x;                      // == feature index f
    __shared__ int sIdx[Kdim];
    __shared__ int sEt[Kdim];
    size_t kbase = ((size_t)(b * Adim + a) * Ndim + n) * Kdim;
    if (tid < Kdim) { sIdx[tid] = nn_idx[kbase + tid]; sEt[tid] = etype[kbase + tid]; }
    __syncthreads();

    const float* feat = nfeature + (size_t)(b * Adim + a) * Ndim * Fdim;
    float a0 = 0.f, a1 = 0.f, a2 = 0.f;
    int c0 = 0, c1 = 0, c2 = 0;
#pragma unroll
    for (int k = 0; k < Kdim; ++k) {
        int e = sEt[k];
        float x = feat[(size_t)sIdx[k] * Fdim + tid];
        if (e == 0)      { a0 += x; c0++; }
        else if (e == 1) { a1 += x; c1++; }
        else             { a2 += x; c2++; }
    }
    size_t rowo = ((size_t)(b * Adim + a) * Ndim + n) * KD;
    agg_bf[rowo + 0 * Fdim + tid] = f32_to_bf16_rne(a0);
    agg_bf[rowo + 1 * Fdim + tid] = f32_to_bf16_rne(a1);
    agg_bf[rowo + 2 * Fdim + tid] = f32_to_bf16_rne(a2);
    if (tid == 0)
        cntpack[(b * Adim + a) * Ndim + n] =
            (unsigned)c0 | ((unsigned)c1 << 8) | ((unsigned)c2 << 16);
}

// ---------------------------------------------------------------------------
// Kernel 3: pw = relu( (agg @ W'[a] + cnt.b) / 16 )
//   GEMM [16 n-rows x 768] @ [768 x 256] per block, WMMA bf16, f32 accum.
// ---------------------------------------------------------------------------
__global__ __launch_bounds__(256)
void pw_gemm_kernel(const unsigned short* __restrict__ agg_bf,
                    const unsigned short* __restrict__ W_bf,     // [A,3F,F]
                    const float*          __restrict__ bias,     // [A,E,F]
                    const unsigned int*   __restrict__ cntpack,
                    float*          __restrict__ pw_f32,
                    unsigned short* __restrict__ pw_bf) {
    int n0 = blockIdx.x * 16, a = blockIdx.y, b = blockIdx.z;
    int tid = threadIdx.x, lane = tid & 31, wave = tid >> 5;
    int hi = lane >> 4, lid = lane & 15;

    __shared__ unsigned int shA[16 * 384];   // 16 rows x 768 bf16 = 24 KB
    size_t slice = (size_t)(b * Adim + a) * Ndim;

    for (int r = 0; r < 16; ++r) {
        int n = n0 + r;
        bool ok = (n < Ndim);
        const unsigned int* src = (const unsigned int*)(agg_bf + (slice + n) * KD);
        for (int c = tid; c < 384; c += 256) shA[r * 384 + c] = ok ? src[c] : 0u;
    }
    __syncthreads();

    v8f z = {};
    v8f acc[2] = {z, z};
    const unsigned short* Wp = W_bf + (size_t)a * KD * Fdim;

    for (int kb = 0; kb < KD; kb += 32) {
        Frag16 af;
        int abase = lid * 384 + (kb >> 1) + hi * 4;
#pragma unroll
        for (int j = 0; j < 4; ++j) {
            af.u[j]     = shA[abase + j];       // K = kb + 2j   (+8 for hi lanes)
            af.u[j + 4] = shA[abase + 8 + j];   // K = kb+16+2j  (+8 for hi lanes)
        }
#pragma unroll
        for (int t = 0; t < 2; ++t) {
            int colbase = (wave * 2 + t) * 16;
            const unsigned short* bp =
                (Wp + (size_t)(kb + hi * 8) * Fdim + colbase + lid);
            Frag16 bf_;
#pragma unroll
            for (int j = 0; j < 8; ++j) { int kk = (j & 3) * 2 + (j >> 2) * 16; bf_.u[j] = (unsigned)bp[(size_t)kk * Fdim] | ((unsigned)bp[(size_t)(kk + 1) * Fdim] << 16); }
            acc[t] = __builtin_amdgcn_wmma_f32_16x16x32_bf16(
                false, af.v, false, bf_.v, (short)0, acc[t], false, false);
        }
    }

#pragma unroll
    for (int t = 0; t < 2; ++t) {
        int col = (wave * 2 + t) * 16 + lid;
        float b0  = bias[(a * Edim + 0) * Fdim + col];
        float b1  = bias[(a * Edim + 1) * Fdim + col];
        float b2v = bias[(a * Edim + 2) * Fdim + col];
#pragma unroll
        for (int r = 0; r < 8; ++r) {
            int n = n0 + r + hi * 8;
            if (n < Ndim) {
                unsigned cp = cntpack[slice + n];
                float v = acc[t][r]
                        + (float)(cp & 31u) * b0
                        + (float)((cp >> 8) & 31u) * b1
                        + (float)((cp >> 16) & 31u) * b2v;
                v *= (1.0f / (float)Kdim);
                v = v > 0.f ? v : 0.f;
                size_t o = (slice + n) * Fdim + col;
                pw_f32[o] = v;
                pw_bf[o]  = f32_to_bf16_rne(v);
            }
        }
    }
}

// ---------------------------------------------------------------------------
// Kernel 4: mutual[b,n,f] = sum_a pw[b,a,n,f]  (bf16 out)
// ---------------------------------------------------------------------------
__global__ void mutual_kernel(const float* __restrict__ pw_f32,
                              unsigned short* __restrict__ mu_bf) {
    int i = blockIdx.x * blockDim.x + threadIdx.x;
    if (i >= Bdim * Ndim * Fdim) return;
    int f = i & (Fdim - 1);
    int n = (i / Fdim) % Ndim;
    int b = i / (Fdim * Ndim);
    float s = 0.f;
#pragma unroll
    for (int a = 0; a < Adim; ++a)
        s += pw_f32[((size_t)(b * Adim + a) * Ndim + n) * Fdim + f];
    mu_bf[i] = f32_to_bf16_rne(s);
}

// ---------------------------------------------------------------------------
// Kernel 5: out = relu(pw@W1[a] + mutual@W2[a] + b2[a]) + nfeature
// ---------------------------------------------------------------------------
__global__ __launch_bounds__(256)
void hop_kernel(const unsigned short* __restrict__ pw_bf,
                const unsigned short* __restrict__ mu_bf,
                const unsigned short* __restrict__ W1_bf,  // [A,F,F]
                const unsigned short* __restrict__ W2_bf,  // [A,F,F]
                const float* __restrict__ b2,              // [A,F]
                const float* __restrict__ nfeature,
                float* __restrict__ out) {
    int n0 = blockIdx.x * 16, a = blockIdx.y, b = blockIdx.z;
    int tid = threadIdx.x, lane = tid & 31, wave = tid >> 5;
    int hi = lane >> 4, lid = lane & 15;

    __shared__ unsigned int shP[16 * 128];
    __shared__ unsigned int shM[16 * 128];
    size_t slice = (size_t)(b * Adim + a) * Ndim;

    for (int r = 0; r < 16; ++r) {
        int n = n0 + r;
        bool ok = (n < Ndim);
        const unsigned int* sp = (const unsigned int*)(pw_bf + (slice + n) * Fdim);
        const unsigned int* sm = (const unsigned int*)(mu_bf + ((size_t)b * Ndim + n) * Fdim);
        for (int c = tid; c < 128; c += 256) {
            shP[r * 128 + c] = ok ? sp[c] : 0u;
            shM[r * 128 + c] = ok ? sm[c] : 0u;
        }
    }
    __syncthreads();

    v8f z = {};
    v8f acc[2] = {z, z};
    const unsigned short* W1p = W1_bf + (size_t)a * Fdim * Fdim;
    const unsigned short* W2p = W2_bf + (size_t)a * Fdim * Fdim;

    for (int kb = 0; kb < Fdim; kb += 32) {
        Frag16 aP, aM;
        int abase = lid * 128 + (kb >> 1) + hi * 4;
#pragma unroll
        for (int j = 0; j < 4; ++j) {
            aP.u[j] = shP[abase + j];     aP.u[j + 4] = shP[abase + 8 + j];
            aM.u[j] = shM[abase + j];     aM.u[j + 4] = shM[abase + 8 + j];
        }
#pragma unroll
        for (int t = 0; t < 2; ++t) {
            int colbase = (wave * 2 + t) * 16;
            const unsigned short* p1 =
                (W1p + (size_t)(kb + hi * 8) * Fdim + colbase + lid);
            const unsigned short* p2 =
                (W2p + (size_t)(kb + hi * 8) * Fdim + colbase + lid);
            Frag16 f1, f2;
#pragma unroll
            for (int j = 0; j < 8; ++j) { int kk = (j & 3) * 2 + (j >> 2) * 16; f1.u[j] = (unsigned)p1[(size_t)kk * Fdim] | ((unsigned)p1[(size_t)(kk + 1) * Fdim] << 16); f2.u[j] = (unsigned)p2[(size_t)kk * Fdim] | ((unsigned)p2[(size_t)(kk + 1) * Fdim] << 16); }
            acc[t] = __builtin_amdgcn_wmma_f32_16x16x32_bf16(
                false, aP.v, false, f1.v, (short)0, acc[t], false, false);
            acc[t] = __builtin_amdgcn_wmma_f32_16x16x32_bf16(
                false, aM.v, false, f2.v, (short)0, acc[t], false, false);
        }
    }

#pragma unroll
    for (int t = 0; t < 2; ++t) {
        int col = (wave * 2 + t) * 16 + lid;
        float bb = b2[a * Fdim + col];
#pragma unroll
        for (int r = 0; r < 8; ++r) {
            int n = n0 + r + hi * 8;
            if (n < Ndim) {
                float v = acc[t][r] + bb;
                v = v > 0.f ? v : 0.f;
                size_t o = (slice + n) * Fdim + col;
                out[o] = v + nfeature[o];
            }
        }
    }
}

// ---------------------------------------------------------------------------
extern "C" void kernel_launch(void* const* d_in, const int* in_sizes, int n_in,
                              void* d_out, int out_size, void* d_ws, size_t ws_size,
                              hipStream_t stream) {
    (void)in_sizes; (void)n_in; (void)out_size; (void)ws_size;

    const float* nfeature = (const float*)d_in[0];   // [B,A,N,F]
    const int*   nn_idx   = (const int*)d_in[1];     // [B,A,N,K]
    const int*   etype    = (const int*)d_in[2];     // [B,A,N,K]
    const float* W        = (const float*)d_in[3];   // [A,E,F,F]
    const float* bias     = (const float*)d_in[4];   // [A,E,F]
    const float* W1       = (const float*)d_in[5];   // [A,F,F]
    const float* W2       = (const float*)d_in[6];   // [A,F,F]
    const float* b2       = (const float*)d_in[7];   // [A,F]
    float* out = (float*)d_out;                      // [B,A,N,F]

    // workspace layout (~35 MB total)
    unsigned short* W_bf    = (unsigned short*)d_ws;             // A*E*F*F   = 983040
    unsigned short* W1_bf   = W_bf  + (size_t)Adim * Edim * Fdim * Fdim;
    unsigned short* W2_bf   = W1_bf + (size_t)Adim * Fdim * Fdim;
    unsigned short* agg_bf  = W2_bf + (size_t)Adim * Fdim * Fdim; // B*A*N*3F = 7,680,000
    unsigned int*   cntpack = (unsigned int*)(agg_bf + (size_t)Bdim * Adim * Ndim * KD);
    float*          pw_f32  = (float*)(cntpack + (size_t)Bdim * Adim * Ndim);
    unsigned short* pw_bf   = (unsigned short*)(pw_f32 + (size_t)Bdim * Adim * Ndim * Fdim);
    unsigned short* mu_bf   = pw_bf + (size_t)Bdim * Adim * Ndim * Fdim;

    const int nW  = Adim * Edim * Fdim * Fdim;   // 983040
    const int nW1 = Adim * Fdim * Fdim;          // 327680

    cvt_bf16_kernel<<<(nW  + 255) / 256, 256, 0, stream>>>(W,  W_bf,  nW);
    cvt_bf16_kernel<<<(nW1 + 255) / 256, 256, 0, stream>>>(W1, W1_bf, nW1);
    cvt_bf16_kernel<<<(nW1 + 255) / 256, 256, 0, stream>>>(W2, W2_bf, nW1);

    agg_kernel<<<dim3(Ndim, Adim, Bdim), Fdim, 0, stream>>>(
        nfeature, nn_idx, etype, agg_bf, cntpack);

    pw_gemm_kernel<<<dim3((Ndim + 15) / 16, Adim, Bdim), 256, 0, stream>>>(
        agg_bf, W_bf, bias, cntpack, pw_f32, pw_bf);

    mutual_kernel<<<(Bdim * Ndim * Fdim + 255) / 256, 256, 0, stream>>>(pw_f32, mu_bf);

    hop_kernel<<<dim3((Ndim + 15) / 16, Adim, Bdim), 256, 0, stream>>>(
        pw_bf, mu_bf, W1_bf, W2_bf, b2, nfeature, out);
}
